// GraphMemoryVQ_24902220382594
// MI455X (gfx1250) — matrix operand order repaired and checked by reading, not compiled, not run on hardware
//
#include <hip/hip_runtime.h>
#include <hip/hip_bf16.h>

// ---------------------------------------------------------------------------
// VQ codebook lookup for MI455X (gfx1250, wave32, WMMA bf16 16x16x32).
//
// S = Z @ C^T (M=16384, N=8192, K=1024, 0.275 TFLOP); fused argmin, gather,
// loss. Codebook (16MB bf16) is L2-resident; traffic scales with M/M_tile,
// so A (full-K bf16 panel of 128 tokens, 258KB) lives in LDS for the whole
// kernel and B is streamed in double-buffered 64x128 K-panels via
// GLOBAL_LOAD_ASYNC_TO_LDS_B128 (ASYNCcnt), overlapping staging with WMMA
// compute. Distances: f32-accumulating bf16 WMMA; z_q / loss are exact f32.
// ---------------------------------------------------------------------------

typedef __attribute__((ext_vector_type(8)))  float   v8f;
typedef __attribute__((ext_vector_type(16))) __bf16  v16bf;
typedef __attribute__((ext_vector_type(8)))  __bf16  v8bf;
typedef __attribute__((ext_vector_type(4)))  __bf16  v4bf;
typedef __attribute__((ext_vector_type(2)))  __bf16  v2bf;
typedef __attribute__((ext_vector_type(4)))  int     v4i;

#ifndef __has_builtin
#define __has_builtin(x) 0
#endif
#if __has_builtin(__builtin_amdgcn_global_load_async_to_lds_b128)
#define HAVE_ASYNC_LDS 1
typedef __attribute__((address_space(1))) v4i* as1_v4i_p;   // global int4*
typedef __attribute__((address_space(3))) v4i* as3_v4i_p;   // LDS int4*
#else
#define HAVE_ASYNC_LDS 0
#endif

#define N_TOKENS 16384
#define LATENT   512
#define KDIM     1024
#define NSYM     8192

#define MT       128          // tokens per workgroup
#define NC       64           // codes per N-chunk
#define BK       128          // K-panel of B staged in LDS (double buffered)
#define A_STRIDE 1032         // bf16/row: 1024 + 8 pad (16B-aligned rows, bank-spread)
#define B_STRIDE 136          // bf16/row:  128 + 8 pad
#define NPANEL   (KDIM / BK)  // 8 K-panels per chunk
#define NCHUNK   (NSYM / NC)  // 128 chunks

#define WMMA_BF16(A_, B_, C_) \
  __builtin_amdgcn_wmma_f32_16x16x32_bf16(false, (A_), false, (B_), (short)0, (C_), false, false)

static __device__ __forceinline__ float clip5(float x) {
  return fminf(fmaxf(x, -5.0f), 5.0f);
}

// --------------------------- prep: codebook -> bf16, cnorm ------------------
__global__ __launch_bounds__(256) void vq_prep_codebook(
    const float* __restrict__ cb, __bf16* __restrict__ cbh,
    float* __restrict__ cnorm) {
  const int wave = threadIdx.x >> 5;
  const int lane = threadIdx.x & 31;
  const int row  = blockIdx.x * 8 + wave;                 // 8 rows / block
  const float*  src = cb  + (size_t)row * KDIM;
  __bf16*       dst = cbh + (size_t)row * KDIM;
  float ss = 0.0f;
#pragma unroll
  for (int i = 0; i < 8; ++i) {
    const int c = i * 128 + lane * 4;
    float4 v = *(const float4*)(src + c);
    ss += v.x * v.x + v.y * v.y + v.z * v.z + v.w * v.w;
    v4bf b;
    b[0] = (__bf16)v.x; b[1] = (__bf16)v.y; b[2] = (__bf16)v.z; b[3] = (__bf16)v.w;
    *(v4bf*)(dst + c) = b;
  }
#pragma unroll
  for (int m = 16; m > 0; m >>= 1) ss += __shfl_xor(ss, m, 32);
  if (lane == 0) cnorm[row] = ss;
}

// ------------------------------- main kernel --------------------------------
// 256 threads = 8 waves, arranged 4(M) x 2(N). Each wave: 32M x 32N per chunk
// = 2x2 WMMA tiles, v8f accumulators. 128 N-chunks cover all 8192 codes.
__global__ __launch_bounds__(256, 1) void vq_main(
    const float*  __restrict__ zr,
    const float*  __restrict__ zi,
    const float*  __restrict__ cb,      // original f32 codebook (exact gather)
    const __bf16* __restrict__ cbh,
    const float*  __restrict__ cnorm,
    float* __restrict__ out_zqr,
    float* __restrict__ out_zqi,
    float* __restrict__ out_idx,
    float* __restrict__ partials) {

  __shared__ __bf16 ldsA[MT * A_STRIDE];             // 264192 B
  __shared__ __bf16 ldsB[2 * NC * B_STRIDE];         //  34816 B (double buffer)
  __shared__ unsigned long long ldsKey[MT];          //   1024 B
  __shared__ float ldsRed[256];                      //   1024 B

  const int tid     = threadIdx.x;
  const int rowbase = blockIdx.x * MT;

  // ---- stage A panel (clip + f32->bf16) once; init argmin keys ----
  for (int r = tid; r < MT; r += 256) ldsKey[r] = ~0ull;
  for (int p = tid; p < MT * (KDIM / 2); p += 256) {
    const int r = p >> 9;                 // 512 float-pairs per row
    const int k = (p & 511) * 2;          // even k: pair never straddles 512
    const float* src = (k < LATENT)
        ? (zr + (size_t)(rowbase + r) * LATENT + k)
        : (zi + (size_t)(rowbase + r) * LATENT + (k - LATENT));
    float2 v = *(const float2*)src;
    v2bf b;
    b[0] = (__bf16)clip5(v.x);
    b[1] = (__bf16)clip5(v.y);
    *(v2bf*)&ldsA[r * A_STRIDE + k] = b;
  }

  const int lane  = tid & 31;
  const int w     = tid >> 5;
  const int wm    = w & 3;                // 4 M-groups of 32 rows
  const int wn    = w >> 2;               // 2 N-groups of 32 cols
  const int lhalf = lane >> 4;
  const int l15   = lane & 15;

  // A fragment rows (lane = M within 16-row subtile); K-offset per half-wave.
  const int arow0 = (wm * 32 + l15) * A_STRIDE;
  const int arow1 = arow0 + 16 * A_STRIDE;
  const int aoff  = lhalf * 8;            // low lanes K0..7/16..23, high K8..15/24..31
  // B fragment rows (lane = N); low lanes K0..7/16..23, high K8..15/24..31.
  const int brow0 = (wn * 32 + l15) * B_STRIDE + lhalf * 8;
  const int brow1 = brow0 + 16 * B_STRIDE;

  // B-panel staging: thread -> (code row r, 64B segment q); 4x b128 per thread.
  const int sr = tid >> 2;
  const int sq = tid & 3;
  auto stageB = [&](int chnk, int kb, int b) {
    const __bf16* src = cbh + (size_t)(chnk * NC + sr) * KDIM + kb + sq * 32;
    __bf16*       dst = &ldsB[b * (NC * B_STRIDE) + sr * B_STRIDE + sq * 32];
#if HAVE_ASYNC_LDS
#pragma unroll
    for (int i = 0; i < 4; ++i)
      __builtin_amdgcn_global_load_async_to_lds_b128(
          (as1_v4i_p)(src + i * 8), (as3_v4i_p)(dst + i * 8), 0, 0);
#else
    const uint4* s4 = (const uint4*)src;
    uint4*       d4 = (uint4*)dst;
#pragma unroll
    for (int i = 0; i < 4; ++i) d4[i] = s4[i];
    __builtin_prefetch(src + BK, 0, 1);
#endif
  };

  float bestV[2][8];
  int   bestI[2][8];
#pragma unroll
  for (int ms = 0; ms < 2; ++ms)
#pragma unroll
    for (int v = 0; v < 8; ++v) { bestV[ms][v] = __builtin_inff(); bestI[ms][v] = 0; }

  int buf = 0;
  stageB(0, 0, 0);                        // prologue: first panel in flight

#pragma unroll 1
  for (int chunk = 0; chunk < NCHUNK; ++chunk) {
    v8f acc[2][2];
#pragma unroll
    for (int ms = 0; ms < 2; ++ms)
#pragma unroll
      for (int ns = 0; ns < 2; ++ns) acc[ms][ns] = (v8f){0, 0, 0, 0, 0, 0, 0, 0};

#pragma unroll 1
    for (int kbi = 0; kbi < NPANEL; ++kbi) {
#if HAVE_ASYNC_LDS
      asm volatile("s_wait_asynccnt 0x0" ::: "memory");  // our panel writes done
#endif
      __syncthreads();                                   // all waves' writes visible
      // Stage the next panel into the other buffer (overlaps with compute).
      if (kbi + 1 < NPANEL)            stageB(chunk, (kbi + 1) * BK, buf ^ 1);
      else if (chunk + 1 < NCHUNK)     stageB(chunk + 1, 0, buf ^ 1);

      const __bf16* Bb = ldsB + buf * (NC * B_STRIDE);
#pragma unroll
      for (int ks = 0; ks < BK; ks += 32) {
        const int ka = kbi * BK + ks + aoff;
        v8bf a0lo = *(const v8bf*)&ldsA[arow0 + ka];
        v8bf a0hi = *(const v8bf*)&ldsA[arow0 + ka + 16];
        v8bf a1lo = *(const v8bf*)&ldsA[arow1 + ka];
        v8bf a1hi = *(const v8bf*)&ldsA[arow1 + ka + 16];
        v16bf A0 = __builtin_shufflevector(a0lo, a0hi, 0,1,2,3,4,5,6,7,8,9,10,11,12,13,14,15);
        v16bf A1 = __builtin_shufflevector(a1lo, a1hi, 0,1,2,3,4,5,6,7,8,9,10,11,12,13,14,15);
        v8bf b0lo = *(const v8bf*)&Bb[brow0 + ks];
        v8bf b0hi = *(const v8bf*)&Bb[brow0 + ks + 16];
        v8bf b1lo = *(const v8bf*)&Bb[brow1 + ks];
        v8bf b1hi = *(const v8bf*)&Bb[brow1 + ks + 16];
        v16bf B0 = __builtin_shufflevector(b0lo, b0hi, 0,1,2,3,4,5,6,7,8,9,10,11,12,13,14,15);
        v16bf B1 = __builtin_shufflevector(b1lo, b1hi, 0,1,2,3,4,5,6,7,8,9,10,11,12,13,14,15);
        acc[0][0] = WMMA_BF16(A0, B0, acc[0][0]);
        acc[0][1] = WMMA_BF16(A0, B1, acc[0][1]);
        acc[1][0] = WMMA_BF16(A1, B0, acc[1][0]);
        acc[1][1] = WMMA_BF16(A1, B1, acc[1][1]);
      }
      buf ^= 1;
    }

    // ---- fold this chunk into per-lane running argmin ----
    const int jbase = chunk * NC + wn * 32;
#pragma unroll
    for (int ns = 0; ns < 2; ++ns) {
      const int j = jbase + ns * 16 + l15;     // lane = N in C/D layout
      const float cn = cnorm[j];
#pragma unroll
      for (int ms = 0; ms < 2; ++ms)
#pragma unroll
        for (int v = 0; v < 8; ++v) {
          const float dv = fmaf(-2.0f, acc[ms][ns][v], cn);   // ||z||^2 const per row
          if (dv < bestV[ms][v]) { bestV[ms][v] = dv; bestI[ms][v] = j; }
        }
    }
  }

  // ---- cross-lane / cross-wave argmin via ds_min_u64 on ordered keys ----
#pragma unroll
  for (int ms = 0; ms < 2; ++ms)
#pragma unroll
    for (int v = 0; v < 8; ++v) {
      const int row = wm * 32 + ms * 16 + v + 8 * lhalf;      // C/D: M = v + 8*half
      unsigned int u = __float_as_uint(bestV[ms][v]);
      u = (u & 0x80000000u) ? ~u : (u | 0x80000000u);         // monotonic float order
      const unsigned long long key =
          ((unsigned long long)u << 32) | (unsigned int)bestI[ms][v];
      atomicMin(&ldsKey[row], key);                           // min dist, then min idx
    }
  __syncthreads();

  // ---- gather z_q (exact f32), emit outputs, accumulate loss ----
  float lsum = 0.0f;
  for (int p = tid; p < MT * KDIM; p += 256) {
    const int r = p >> 10, k = p & 1023;
    const int g = rowbase + r;
    const int idx = (int)(ldsKey[r] & 0xFFFFFFFFull);
    const float cv = cb[(size_t)idx * KDIM + k];
    float zf = (k < LATENT) ? zr[(size_t)g * LATENT + k]
                            : zi[(size_t)g * LATENT + (k - LATENT)];
    zf = clip5(zf);
    const float d = cv - zf;
    lsum = fmaf(d, d, lsum);
    if (k < LATENT) out_zqr[(size_t)g * LATENT + k] = cv;
    else            out_zqi[(size_t)g * LATENT + (k - LATENT)] = cv;
  }
  if (tid < MT)
    out_idx[rowbase + tid] = (float)(unsigned int)(ldsKey[tid] & 0xFFFFFFFFull);

  ldsRed[tid] = lsum;
  __syncthreads();
#pragma unroll
  for (int s = 128; s > 0; s >>= 1) {
    if (tid < s) ldsRed[tid] += ldsRed[tid + s];
    __syncthreads();
  }
  if (tid == 0) partials[blockIdx.x] = ldsRed[0];
}

// -------- deterministic, ordered loss finalize (graph-replay safe) ----------
__global__ void vq_loss_final(const float* __restrict__ partials,
                              float* __restrict__ loss_out) {
  if (threadIdx.x == 0) {
    float s = 0.0f;
    for (int i = 0; i < NCHUNK; ++i) s += partials[i];
    // loss_vq + 0.25*loss_commit collapses to 1.25 * mean((z_q - z_flat)^2)
    *loss_out = s * (1.25f / ((float)N_TOKENS * (float)KDIM));
  }
}

extern "C" void kernel_launch(void* const* d_in, const int* in_sizes, int n_in,
                              void* d_out, int out_size, void* d_ws, size_t ws_size,
                              hipStream_t stream) {
  const float* zr = (const float*)d_in[0];
  const float* zi = (const float*)d_in[1];
  const float* cb = (const float*)d_in[2];

  float* out   = (float*)d_out;
  float* zqr   = out;                                    // 16384*512
  float* zqi   = out + (size_t)N_TOKENS * LATENT;        // 16384*512
  float* loss  = out + (size_t)2 * N_TOKENS * LATENT;    // 1
  float* idxf  = loss + 1;                               // 16384

  __bf16* cbh      = (__bf16*)d_ws;                                  // 16 MB
  float*  cnorm    = (float*)((char*)d_ws + (size_t)NSYM * KDIM * 2);
  float*  partials = cnorm + NSYM;

  vq_prep_codebook<<<NSYM / 8, 256, 0, stream>>>(cb, cbh, cnorm);
  vq_main<<<N_TOKENS / MT, 256, 0, stream>>>(zr, zi, cb, cbh, cnorm,
                                             zqr, zqi, idxf, partials);
  vq_loss_final<<<1, 32, 0, stream>>>(partials, loss);
}
